// MultiHeadAttention_65515431133614
// MI455X (gfx1250) — matrix operand order repaired and checked by reading, not compiled, not run on hardware
//
#include <hip/hip_runtime.h>

// ---------------------------------------------------------------------------
// MultiHeadAttention on MI455X (gfx1250, wave32, WMMA)
//   x:[B,S,D] f32 -> QKV proj (bf16 WMMA, ping-pong LDS, 1 barrier/step) ->
//   flash attention (async-to-LDS triple-buffered K/V, 1 barrier/tile) ->
//   out proj.   B=2 S=2048 D=1024 H=16 HD=64
// ---------------------------------------------------------------------------

#define DEV static __device__ __forceinline__

typedef __attribute__((ext_vector_type(4)))  float  v4f;
typedef __attribute__((ext_vector_type(8)))  float  v8f;
typedef __attribute__((ext_vector_type(4)))  __bf16 v4bf;
typedef __attribute__((ext_vector_type(8)))  __bf16 v8bf;
typedef __attribute__((ext_vector_type(16))) __bf16 v16bf;

static constexpr int B_  = 2;
static constexpr int S_  = 2048;
static constexpr int D_  = 1024;
static constexpr int H_  = 16;
static constexpr int HD_ = 64;
static constexpr int M_  = B_ * S_;       // 4096 rows for the projection GEMMs

DEV v8f zero8() { v8f z = {0.f,0.f,0.f,0.f,0.f,0.f,0.f,0.f}; return z; }

DEV v8bf ld8bf(const __bf16* p) { return *(const v8bf*)p; }

DEV v16bf join16(v8bf lo, v8bf hi) {
  return __builtin_shufflevector(lo, hi, 0,1,2,3,4,5,6,7,8,9,10,11,12,13,14,15);
}

// D = A(16x32 bf16) * B(32x16 bf16) + C(16x16 f32)
DEV v8f wmma_bf16(v16bf a, v16bf b, v8f c) {
  return __builtin_amdgcn_wmma_f32_16x16x32_bf16(
      /*neg_a=*/false, a, /*neg_b=*/false, b,
      /*c_mod=*/(short)0, c, /*reuse_a=*/false, /*reuse_b=*/false);
}

// --- CDNA5 async copy: global -> LDS, 16 bytes per lane, ASYNCcnt-tracked ---
DEV uint32_t lds_off(const void* p) { return (uint32_t)(uintptr_t)p; }

DEV void async_copy_b128(uint32_t lds_addr, const void* gptr) {
  asm volatile("global_load_async_to_lds_b128 %0, %1, off"
               :: "v"(lds_addr), "v"((uint64_t)(uintptr_t)gptr)
               : "memory");
}
DEV void wait_async_le4() { asm volatile("s_wait_asynccnt 0x4" ::: "memory"); }
DEV void wait_async_0()   { asm volatile("s_wait_asynccnt 0x0" ::: "memory"); }

// ---------------------------------------------------------------------------
// Kernel 1: fused QKV projection.
//   grid = (D/128, M/128, 3), block = 256 (8 waves), blockIdx.z selects W.
//   Block tile 128x128, wave tile 32x64 (2x4 WMMA accs), K-step 32.
//   Ping-pong LDS + register prefetch: ONE barrier per K-step; global loads
//   for tile t+1 and the regs->LDS stage both overlap the 8 WMMAs of tile t.
//   mode 0: Q -> [B,H,S,HD] bf16, pre-scaled by 1/sqrt(HD)
//   mode 1: K -> [B,H,S,HD] bf16
//   mode 2: V -> [B,H,HD,S] bf16 (transposed: PV B-fragments contiguous)
// ---------------------------------------------------------------------------
__global__ __launch_bounds__(256)
void qkv_gemm_kernel(const float* __restrict__ x,
                     const float* __restrict__ Wq, const float* __restrict__ bq,
                     const float* __restrict__ Wk, const float* __restrict__ bk,
                     const float* __restrict__ Wv, const float* __restrict__ bv,
                     __bf16* __restrict__ Q, __bf16* __restrict__ K,
                     __bf16* __restrict__ Vt)
{
  const int mode = blockIdx.z;
  const float* __restrict__ Wm = (mode == 0) ? Wq : (mode == 1 ? Wk : Wv);
  const float* __restrict__ bm = (mode == 0) ? bq : (mode == 1 ? bk : bv);

  __shared__ __bf16 As[2][128][32];   // ping-pong, row-major  [m][k]
  __shared__ __bf16 Bs[2][128][32];   // ping-pong, transposed [n][k]

  const int tid  = threadIdx.x;
  const int lane = tid & 31;
  const int wid  = tid >> 5;
  const int m0   = blockIdx.y * 128;
  const int n0   = blockIdx.x * 128;
  const int wm   = (wid >> 1) * 32;   // 4 waves along M
  const int wn   = (wid & 1) * 64;    // 2 waves along N
  const int row16 = lane & 15;
  const int kg    = lane >> 4;

  // staging coordinates
  const int a_c4 = (tid & 7) * 4,  a_r = tid >> 3;   // A: 32 rows/pass, 4 passes
  const int b_n4 = (tid & 31) * 4, b_k = tid >> 5;   // B: 8 k-rows/pass, 4 passes

  v8f acc[2][4];
  for (int i = 0; i < 2; i++)
    for (int j = 0; j < 4; j++) acc[i][j] = zero8();

  v4f ra[4], rb[4];
  auto loadA = [&](int k0) {
    for (int p = 0; p < 4; p++)
      ra[p] = *(const v4f*)&x[(size_t)(m0 + a_r + p * 32) * D_ + k0 + a_c4];
  };
  auto loadB = [&](int k0) {
    for (int p = 0; p < 4; p++)
      rb[p] = *(const v4f*)&Wm[(size_t)(k0 + b_k + p * 8) * D_ + n0 + b_n4];
  };
  auto stageLDS = [&](int buf) {
    for (int p = 0; p < 4; p++) {
      v4bf o;
      for (int i = 0; i < 4; i++) o[i] = (__bf16)ra[p][i];
      *(v4bf*)&As[buf][a_r + p * 32][a_c4] = o;
    }
    for (int p = 0; p < 4; p++)
      for (int i = 0; i < 4; i++)
        Bs[buf][b_n4 + i][b_k + p * 8] = (__bf16)rb[p][i];
  };

  constexpr int NT = D_ / 32;          // 32 K-steps
  loadA(0);
  loadB(0);
  stageLDS(0);
  for (int it = 0; it < NT; it++) {
    const int buf = it & 1;
    __syncthreads();                   // buf's staging complete (all waves)

    if (it + 1 < NT) {                 // global loads for tile it+1 overlap WMMA
      loadA((it + 1) * 32);
      loadB((it + 1) * 32);
      if (it + 2 < NT) {               // L2 prefetch one more tile ahead
        __builtin_prefetch(&x[(size_t)(m0 + a_r) * D_ + (it + 2) * 32 + a_c4], 0, 1);
        __builtin_prefetch(&Wm[(size_t)((it + 2) * 32 + b_k) * D_ + n0 + b_n4], 0, 1);
      }
    }

    v16bf af[2];
    for (int mi = 0; mi < 2; mi++) {
      const __bf16* base = &As[buf][wm + mi * 16 + row16][0];
      // A 16x32 layout: lane row = l%16, K = {kg*8..+7} U {16+kg*8..+7}
      af[mi] = join16(ld8bf(base + kg * 8), ld8bf(base + 16 + kg * 8));
    }
    for (int ni = 0; ni < 4; ni++) {
      const __bf16* base = &Bs[buf][wn + ni * 16 + row16][0];
      // B 32x16 layout: lane col = l%16, K = {kg*8..+7} U {16+kg*8..+7}
      v16bf bfr = join16(ld8bf(base + kg * 8), ld8bf(base + 16 + kg * 8));
      for (int mi = 0; mi < 2; mi++)
        acc[mi][ni] = wmma_bf16(af[mi], bfr, acc[mi][ni]);
    }

    if (it + 1 < NT)
      stageLDS(buf ^ 1);               // safe: all waves are past barrier `it`
  }

  // --- epilogue: bias, layout-aware scatter ---
  const int g = lane >> 4;
  for (int ni = 0; ni < 4; ni++) {
    const int n  = n0 + wn + ni * 16 + row16;
    const float bias = bm[n];
    const int h  = n >> 6;     // n / HD
    const int hd = n & 63;     // n % HD
    for (int mi = 0; mi < 2; mi++) {
      for (int v = 0; v < 8; v++) {
        const int m = m0 + wm + mi * 16 + v + g * 8;
        const int b = m >> 11;       // m / S
        const int s = m & 2047;      // m % S
        const float val = acc[mi][ni][v] + bias;
        if (mode == 0) {
          Q[((size_t)(b * H_ + h) * S_ + s) * HD_ + hd] = (__bf16)(val * 0.125f);
        } else if (mode == 1) {
          K[((size_t)(b * H_ + h) * S_ + s) * HD_ + hd] = (__bf16)val;
        } else {
          Vt[((size_t)(b * H_ + h) * HD_ + hd) * S_ + s] = (__bf16)val;
        }
      }
    }
  }
}

// ---------------------------------------------------------------------------
// Kernel 2: causal flash attention, online softmax.
//   grid = (S/64, B*H), block = 128 (4 waves). Each wave owns 16 query rows.
//   K/V tiles (32 keys) staged once per BLOCK into TRIPLE-buffered LDS via
//   CDNA5 global_load_async_to_lds_b128. One barrier per tile: with skew <= 1
//   iteration, staging target (kt+2)%3 never collides with readers of kt%3 or
//   (kt+1)%3. ASYNCcnt: issue 4 ops for kt+1, wait <=4 -> tile kt complete.
// ---------------------------------------------------------------------------
__global__ __launch_bounds__(128)
void attn_kernel(const __bf16* __restrict__ Q, const __bf16* __restrict__ K,
                 const __bf16* __restrict__ Vt, __bf16* __restrict__ attn)
{
  __shared__ __bf16 Ks[3][32][64];  // [buf][key][hd]  4KB per buffer
  __shared__ __bf16 Vs[3][64][32];  // [buf][hd][key]  4KB per buffer
  __shared__ __bf16 P[4][16][32];   // per-wave P tile (C-layout -> A-layout)

  const int tid   = threadIdx.x;
  const int lane  = tid & 31;
  const int w     = tid >> 5;
  const int bh    = blockIdx.y;                  // b*H + h
  const int q0blk = blockIdx.x * 64;
  const int q0    = q0blk + w * 16;              // this wave's query base
  const int row16 = lane & 15;
  const int kg    = lane >> 4;
  const int g     = kg;

  const __bf16* __restrict__ Qb = Q  + (size_t)bh * S_ * HD_;
  const __bf16* __restrict__ Kb = K  + (size_t)bh * S_ * HD_;
  const __bf16* __restrict__ Vb = Vt + (size_t)bh * HD_ * S_;

  // async stage of one 32-key K/V tile: 128 threads x 32B each per matrix
  const int sk_key = tid >> 2, sk_c = (tid & 3) * 16;   // K: [key][hd-chunk]
  const int sv_hd  = tid >> 1, sv_c = (tid & 1) * 16;   // V: [hd][key-chunk]
  auto stage_tile = [&](int kt, int buf) {
    const __bf16* gk = Kb + (size_t)(kt * 32 + sk_key) * HD_ + sk_c;
    uint32_t lk = lds_off(&Ks[buf][sk_key][sk_c]);
    async_copy_b128(lk,      gk);
    async_copy_b128(lk + 16, gk + 8);
    const __bf16* gv = Vb + (size_t)sv_hd * S_ + kt * 32 + sv_c;
    uint32_t lv = lds_off(&Vs[buf][sv_hd][sv_c]);
    async_copy_b128(lv,      gv);
    async_copy_b128(lv + 16, gv + 8);
  };

  // Q A-fragments: held in registers for the whole kv loop
  v16bf qf[2];
  {
    const __bf16* qr = Qb + (size_t)(q0 + row16) * HD_;
    for (int c = 0; c < 2; c++)
      qf[c] = join16(ld8bf(qr + c * 32 + kg * 8),
                     ld8bf(qr + c * 32 + 16 + kg * 8));
  }

  v8f acc[4];
  float mrow[8], lrow[8];
  for (int nb = 0; nb < 4; nb++) acc[nb] = zero8();
  for (int v = 0; v < 8; v++) { mrow[v] = -3.0e38f; lrow[v] = 0.f; }

  const int nkt = (q0blk + 63) / 32 + 1;    // block-level causal tile count
  stage_tile(0, 0);
  for (int kt = 0; kt < nkt; kt++) {
    const int buf = kt % 3;
    const int kb0 = kt * 32;

    if (kt + 1 < nkt) {
      stage_tile(kt + 1, (kt + 1) % 3);   // prefetch next tile (other buffer)
      wait_async_le4();                   // tile kt's 4 async ops completed
    } else {
      wait_async_0();
    }
    __syncthreads();                      // all waves' copies visible

    if (kb0 <= q0 + 15) {                 // wave-uniform causal skip
      // ---- scores: two 16-key sub-tiles, K-dim = HD = 64 (2 chunks) ----
      v8f sc[2];
      for (int j = 0; j < 2; j++) {
        sc[j] = zero8();
        const __bf16* kr = &Ks[buf][j * 16 + row16][0];
        for (int c = 0; c < 2; c++) {
          v16bf kf = join16(ld8bf(kr + c * 32 + kg * 8),
                            ld8bf(kr + c * 32 + 16 + kg * 8));
          sc[j] = wmma_bf16(qf[c], kf, sc[j]);
        }
      }

      // ---- causal mask (only near the diagonal; uniform branch) ----
      if (kb0 + 31 > q0) {
        for (int j = 0; j < 2; j++)
          for (int v = 0; v < 8; v++) {
            const int qi = q0 + v + g * 8;
            const int ki = kb0 + j * 16 + row16;
            if (ki > qi) sc[j][v] = -3.0e38f;
          }
      }

      // ---- online softmax (row stats via xor-shuffles, 16-lane groups) ----
      float scale[8];
      for (int v = 0; v < 8; v++) {
        float t = fmaxf(sc[0][v], sc[1][v]);
        for (int off = 1; off < 16; off <<= 1)
          t = fmaxf(t, __shfl_xor(t, off, 32));
        const float mnew = fmaxf(mrow[v], t);
        scale[v] = __expf(mrow[v] - mnew);
        mrow[v]  = mnew;
        float rs = 0.f;
        for (int j = 0; j < 2; j++) {
          sc[j][v] = __expf(sc[j][v] - mnew);
          rs += sc[j][v];
        }
        for (int off = 1; off < 16; off <<= 1)
          rs += __shfl_xor(rs, off, 32);
        lrow[v] = lrow[v] * scale[v] + rs;
      }

      // ---- P: C-layout regs -> LDS -> A-layout fragment ----
      for (int j = 0; j < 2; j++)
        for (int v = 0; v < 8; v++)
          P[w][v + g * 8][j * 16 + row16] = (__bf16)sc[j][v];

      for (int nb = 0; nb < 4; nb++)
        for (int v = 0; v < 8; v++) acc[nb][v] *= scale[v];

      v16bf pa;
      {
        const __bf16* pr = &P[w][row16][0];
        pa = join16(ld8bf(pr + kg * 8), ld8bf(pr + 16 + kg * 8));
      }

      // ---- PV: V tile is [hd][key] in LDS -> contiguous B-fragments ----
      for (int nb = 0; nb < 4; nb++) {
        const __bf16* vr = &Vs[buf][nb * 16 + row16][0];
        v16bf vf = join16(ld8bf(vr + kg * 8), ld8bf(vr + 16 + kg * 8));
        acc[nb] = wmma_bf16(pa, vf, acc[nb]);
      }
    }
  }

  // ---- epilogue: normalize, scatter to attn [B,S,D] bf16 ----
  const int b = bh >> 4, h = bh & 15;
  for (int nb = 0; nb < 4; nb++) {
    const int hd = nb * 16 + row16;
    for (int v = 0; v < 8; v++) {
      const int q = q0 + v + g * 8;
      const float o = acc[nb][v] / lrow[v];
      attn[((size_t)(b * S_ + q)) * D_ + h * HD_ + hd] = (__bf16)o;
    }
  }
}

// ---------------------------------------------------------------------------
// Kernel 3: output projection: attn(bf16)[4096,1024] @ Wo + bo -> f32 out.
// Same single-barrier ping-pong pipeline as kernel 1; A is already bf16.
// ---------------------------------------------------------------------------
__global__ __launch_bounds__(256)
void out_gemm_kernel(const __bf16* __restrict__ A,
                     const float* __restrict__ Wo, const float* __restrict__ bo,
                     float* __restrict__ out)
{
  __shared__ __bf16 As[2][128][32];
  __shared__ __bf16 Bs[2][128][32];

  const int tid  = threadIdx.x;
  const int lane = tid & 31;
  const int wid  = tid >> 5;
  const int m0   = blockIdx.y * 128;
  const int n0   = blockIdx.x * 128;
  const int wm   = (wid >> 1) * 32;
  const int wn   = (wid & 1) * 64;
  const int row16 = lane & 15;
  const int kg    = lane >> 4;

  const int a_c8 = (tid & 3) * 8,  a_r = tid >> 2;   // A: 64 rows/pass, 2 passes
  const int b_n4 = (tid & 31) * 4, b_k = tid >> 5;   // B: 8 k-rows/pass, 4 passes

  v8f acc[2][4];
  for (int i = 0; i < 2; i++)
    for (int j = 0; j < 4; j++) acc[i][j] = zero8();

  v8bf ra[2];
  v4f  rb[4];
  auto loadA = [&](int k0) {
    for (int p = 0; p < 2; p++)
      ra[p] = *(const v8bf*)&A[(size_t)(m0 + a_r + p * 64) * D_ + k0 + a_c8];
  };
  auto loadB = [&](int k0) {
    for (int p = 0; p < 4; p++)
      rb[p] = *(const v4f*)&Wo[(size_t)(k0 + b_k + p * 8) * D_ + n0 + b_n4];
  };
  auto stageLDS = [&](int buf) {
    for (int p = 0; p < 2; p++)
      *(v8bf*)&As[buf][a_r + p * 64][a_c8] = ra[p];
    for (int p = 0; p < 4; p++)
      for (int i = 0; i < 4; i++)
        Bs[buf][b_n4 + i][b_k + p * 8] = (__bf16)rb[p][i];
  };

  constexpr int NT = D_ / 32;
  loadA(0);
  loadB(0);
  stageLDS(0);
  for (int it = 0; it < NT; it++) {
    const int buf = it & 1;
    __syncthreads();

    if (it + 1 < NT) {
      loadA((it + 1) * 32);
      loadB((it + 1) * 32);
      if (it + 2 < NT) {
        __builtin_prefetch(&A[(size_t)(m0 + a_r) * D_ + (it + 2) * 32 + a_c8], 0, 1);
        __builtin_prefetch(&Wo[(size_t)((it + 2) * 32 + b_k) * D_ + n0 + b_n4], 0, 1);
      }
    }

    v16bf af[2];
    for (int mi = 0; mi < 2; mi++) {
      const __bf16* base = &As[buf][wm + mi * 16 + row16][0];
      af[mi] = join16(ld8bf(base + kg * 8), ld8bf(base + 16 + kg * 8));
    }
    for (int ni = 0; ni < 4; ni++) {
      const __bf16* base = &Bs[buf][wn + ni * 16 + row16][0];
      v16bf bfr = join16(ld8bf(base + kg * 8), ld8bf(base + 16 + kg * 8));
      for (int mi = 0; mi < 2; mi++)
        acc[mi][ni] = wmma_bf16(af[mi], bfr, acc[mi][ni]);
    }

    if (it + 1 < NT)
      stageLDS(buf ^ 1);
  }

  const int g = lane >> 4;
  for (int ni = 0; ni < 4; ni++) {
    const int n = n0 + wn + ni * 16 + row16;
    const float bias = bo[n];
    for (int mi = 0; mi < 2; mi++) {
      for (int v = 0; v < 8; v++) {
        const int m = m0 + wm + mi * 16 + v + g * 8;
        out[(size_t)m * D_ + n] = acc[mi][ni][v] + bias;
      }
    }
  }
}

// ---------------------------------------------------------------------------
// Host-side launcher (graph-capture safe: only kernel launches on `stream`).
// Workspace: Q | K | Vt | attn, 8 MB each (bf16), total 32 MB.
// ---------------------------------------------------------------------------
extern "C" void kernel_launch(void* const* d_in, const int* in_sizes, int n_in,
                              void* d_out, int out_size, void* d_ws, size_t ws_size,
                              hipStream_t stream)
{
  (void)in_sizes; (void)n_in; (void)out_size; (void)ws_size;

  const float* x  = (const float*)d_in[0];
  const float* Wq = (const float*)d_in[1];
  const float* bq = (const float*)d_in[2];
  const float* Wk = (const float*)d_in[3];
  const float* bk = (const float*)d_in[4];
  const float* Wv = (const float*)d_in[5];
  const float* bv = (const float*)d_in[6];
  const float* Wo = (const float*)d_in[7];
  const float* bo = (const float*)d_in[8];
  float* out = (float*)d_out;

  char* ws = (char*)d_ws;
  const size_t seg = (size_t)B_ * H_ * S_ * HD_ * sizeof(__bf16);  // 8 MB
  __bf16* Q    = (__bf16*)(ws);
  __bf16* K    = (__bf16*)(ws + seg);
  __bf16* Vt   = (__bf16*)(ws + 2 * seg);
  __bf16* attn = (__bf16*)(ws + 3 * seg);

  dim3 gQKV(D_ / 128, M_ / 128, 3);
  qkv_gemm_kernel<<<gQKV, 256, 0, stream>>>(x, Wq, bq, Wk, bk, Wv, bv, Q, K, Vt);

  dim3 gAttn(S_ / 64, B_ * H_);
  attn_kernel<<<gAttn, 128, 0, stream>>>(Q, K, Vt, attn);

  dim3 gOut(D_ / 128, M_ / 128);
  out_gemm_kernel<<<gOut, 256, 0, stream>>>(attn, Wo, bo, out);
}
